// LSTM_47038481826052
// MI455X (gfx1250) — matrix operand order repaired and checked by reading, not compiled, not run on hardware
//
#include <hip/hip_runtime.h>

typedef __bf16 bf16_t;
typedef __attribute__((ext_vector_type(16))) __bf16 v16bf;
typedef __attribute__((ext_vector_type(8)))  float  v8f;
typedef unsigned int   u32;
typedef unsigned short u16;

#define T_DIM 512
#define B_DIM 64
#define H_DIM 1024
#define L_DIM 2
#define G4    4096   // 4*H
#define TC    64     // time-chunk length
#define LSTR  20     // LDS row stride in dwords (80B = 5x16B, conflict-free)

// ---------------- helpers ----------------

__device__ __forceinline__ u16 f2bf(float f) {
  u32 u = __float_as_uint(f);
  u32 r = (u + 0x7FFFu + ((u >> 16) & 1u)) >> 16;   // round-to-nearest-even
  return (u16)r;
}

// Async DMA one 16B beat from global memory into LDS (per-lane addresses).
// Tracked by ASYNCcnt; bypasses the VGPR round-trip.
__device__ __forceinline__ void async_b128(void* lds_ptr, const void* gptr) {
  const u32 loff = (u32)(uintptr_t)lds_ptr;   // flat->LDS: low 32 bits are the LDS address
  asm volatile("global_load_async_to_lds_b128 %0, %1, off"
               :: "v"(loff), "v"(gptr)
               : "memory");
}
__device__ __forceinline__ void wait_async0() {
  asm volatile("s_wait_asynccnt 0" ::: "memory");
}

// A fragment (16x32 bf16, MxK): lane m (0-15) rows; K split per lane-half.
__device__ __forceinline__ v16bf frag_a(const u32* s, int row0, int lane) {
  const int m = lane & 15, half = lane >> 4;
  const u32* p = s + (row0 + m) * LSTR + half * 4;
  union { v16bf v; u32 u[8]; } r;
#pragma unroll
  for (int j = 0; j < 8; ++j) {
    const int d = (j < 4) ? j : (8 + (j - 4));      // K 0-7 / 16-23 (half shifts +4 dw)
    r.u[j] = p[d];
  }
  return r.v;
}

// B fragment (32x16 bf16, KxN) from an N-major LDS tile (row = N, K contiguous).
__device__ __forceinline__ v16bf frag_b(const u32* s, int col0, int lane) {
  const int n = lane & 15, kh = lane >> 4;
  const u32* p = s + (col0 + n) * LSTR + kh * 4;
  union { v16bf v; u32 u[8]; } r;
#pragma unroll
  for (int j = 0; j < 8; ++j) r.u[j] = p[(j < 4) ? j : (j + 4)];
  return r.v;
}

__device__ __forceinline__ v8f wmma_bf16(v16bf a, v16bf b, v8f c) {
  return __builtin_amdgcn_wmma_f32_16x16x32_bf16(false, a, false, b, (short)0, c,
                                                 false, false);
}

// ---------------- elementwise kernels ----------------

__global__ void cvt_f32_bf16(const float* __restrict__ in, u16* __restrict__ out,
                             size_t n) {
  size_t i = (size_t)blockIdx.x * blockDim.x + threadIdx.x;
  const size_t stride = (size_t)gridDim.x * blockDim.x;
  for (; i < n; i += stride) out[i] = f2bf(in[i]);
}

__global__ void copy_f32(const float* __restrict__ in, float* __restrict__ out,
                         size_t n) {
  size_t i = (size_t)blockIdx.x * blockDim.x + threadIdx.x;
  const size_t stride = (size_t)gridDim.x * blockDim.x;
  for (; i < n; i += stride) out[i] = in[i];
}

// ---------------- input-projection GEMM ----------------
// C(M=TC*B, N=4096) = A(M,1024) * W(4096,1024)^T  (bf16, K contiguous in both).
// Tile 128x128x32, double-buffered LDS filled by async-to-LDS DMA.
__global__ __launch_bounds__(256) void lstm_gemm_xw(const u32* __restrict__ A,
                                                    const u32* __restrict__ W,
                                                    float* __restrict__ C) {
  __shared__ u32 sA[2][128 * LSTR];
  __shared__ u32 sB[2][128 * LSTR];
  const int tid = threadIdx.x;
  const int lane = tid & 31;
  const int wave = tid >> 5;
  const int wm = wave & 3;          // 32-row block
  const int wn = wave >> 2;         // 64-col block
  const int rowBase = blockIdx.x * 128;
  const int colBase = blockIdx.y * 128;

  v8f acc[2][4];
#pragma unroll
  for (int i = 0; i < 2; ++i)
#pragma unroll
    for (int j = 0; j < 4; ++j) acc[i][j] = {};

  const int ldr = tid >> 1;         // 0..127 tile row
  const int kp  = (tid & 1) * 8;    // dword offset within 16-dword K slab
  const u32* ga = A + (size_t)(rowBase + ldr) * 512 + kp;   // K=1024 bf16 = 512 dw
  const u32* gb = W + (size_t)(colBase + ldr) * 512 + kp;
  const int lbase = ldr * LSTR + kp;

  // prologue: DMA slab 0 into buffer 0
  async_b128(&sA[0][lbase], ga);         async_b128(&sA[0][lbase + 4], ga + 4);
  async_b128(&sB[0][lbase], gb);         async_b128(&sB[0][lbase + 4], gb + 4);

  int buf = 0;
  for (int k0 = 0; k0 < 512; k0 += 16) {
    wait_async0();        // this wave's DMA into sX[buf] complete
    __syncthreads();      // everyone's DMA done; everyone's reads of sX[buf^1] done
    if (k0 + 16 < 512) {  // prefetch next slab into the idle buffer
      const u32* na = ga + k0 + 16;
      const u32* nb = gb + k0 + 16;
      async_b128(&sA[buf ^ 1][lbase], na);     async_b128(&sA[buf ^ 1][lbase + 4], na + 4);
      async_b128(&sB[buf ^ 1][lbase], nb);     async_b128(&sB[buf ^ 1][lbase + 4], nb + 4);
    }
    v16bf af[2];
    af[0] = frag_a(sA[buf], wm * 32, lane);
    af[1] = frag_a(sA[buf], wm * 32 + 16, lane);
    v16bf bfr[4];
#pragma unroll
    for (int j = 0; j < 4; ++j) bfr[j] = frag_b(sB[buf], wn * 64 + j * 16, lane);
#pragma unroll
    for (int i = 0; i < 2; ++i)
#pragma unroll
      for (int j = 0; j < 4; ++j) acc[i][j] = wmma_bf16(af[i], bfr[j], acc[i][j]);
    buf ^= 1;
  }

  const int n = lane & 15, half = lane >> 4;
#pragma unroll
  for (int i = 0; i < 2; ++i)
#pragma unroll
    for (int j = 0; j < 4; ++j) {
      float* cp = C + (size_t)(rowBase + wm * 32 + i * 16 + 8 * half) * G4 +
                  (colBase + wn * 64 + j * 16 + n);
#pragma unroll
      for (int r = 0; r < 8; ++r) cp[(size_t)r * G4] = acc[i][j][r];
    }
}

// ---------------- fused recurrent step ----------------
// One block owns h-columns [jcol*16, jcol*16+16) across all 4 gates.
// wave w: gate g = w&3, row-half mh = w>>2 (rows mh*32 .. +32 of B=64).
__global__ __launch_bounds__(256) void lstm_step(
    const float* __restrict__ xp_t,     // (64, 4096) this timestep
    const u16*   __restrict__ hprev,    // (64, 1024) bf16
    const u32*   __restrict__ Wh,       // (4096, 1024) bf16 as dwords
    const float* __restrict__ bxl,      // (4, 1024)
    const float* __restrict__ bhl,      // (4, 1024)
    float* __restrict__ c_state,        // (64, 1024)
    float* __restrict__ h_f32,          // (64, 1024) -> output sequence
    u16*   __restrict__ h_bf16) {       // (64, 1024) -> next-step / next-layer input
  __shared__ u32   sH[2][64 * LSTR];
  __shared__ u32   sW[2][64 * LSTR];
  __shared__ float sZ[4][64][17];

  const int tid = threadIdx.x;
  const int lane = tid & 31;
  const int wave = tid >> 5;
  const int g = wave & 3;
  const int mh = wave >> 2;
  const int jcol = blockIdx.x;          // 0..63

  v8f acc0 = {}, acc1 = {};

  const int row = tid >> 2;             // 0..63
  const int kp  = (tid & 3) * 8;        // bf16 elements within 32-K slab
  const int gg = row >> 4, nn = row & 15;
  const u16* gh = hprev + (size_t)row * H_DIM + kp;
  const u32* gw = Wh + ((size_t)gg * H_DIM + jcol * 16 + nn) * 512 + (kp >> 1);
  const int lbase = row * LSTR + (kp >> 1);

  // prologue: DMA slab 0 into buffer 0
  async_b128(&sH[0][lbase], gh);
  async_b128(&sW[0][lbase], gw);

  int buf = 0;
  for (int k0 = 0; k0 < H_DIM; k0 += 32) {
    wait_async0();
    __syncthreads();
    if (k0 + 32 < H_DIM) {
      async_b128(&sH[buf ^ 1][lbase], gh + k0 + 32);
      async_b128(&sW[buf ^ 1][lbase], gw + (k0 + 32) / 2);
    }
    const v16bf a0 = frag_a(sH[buf], mh * 32, lane);
    const v16bf a1 = frag_a(sH[buf], mh * 32 + 16, lane);
    const v16bf b  = frag_b(sW[buf], g * 16, lane);
    acc0 = wmma_bf16(a0, b, acc0);
    acc1 = wmma_bf16(a1, b, acc1);
    buf ^= 1;
  }

  {
    const int n = lane & 15, half = lane >> 4;
#pragma unroll
    for (int r = 0; r < 8; ++r) {
      sZ[g][mh * 32 + r + 8 * half][n]      = acc0[r];
      sZ[g][mh * 32 + 16 + r + 8 * half][n] = acc1[r];
    }
  }
  __syncthreads();

#pragma unroll
  for (int ii = 0; ii < 4; ++ii) {
    const int idx = tid + 256 * ii;     // 0..1023 over (64 rows x 16 cols)
    const int r = idx >> 4, col = idx & 15;
    const int cg = jcol * 16 + col;
    const size_t xb = (size_t)r * G4 + cg;
    const float zi = sZ[0][r][col] + xp_t[xb]        + bxl[cg]        + bhl[cg];
    const float zf = sZ[1][r][col] + xp_t[xb + 1024] + bxl[1024 + cg] + bhl[1024 + cg];
    const float zg = sZ[2][r][col] + xp_t[xb + 2048] + bxl[2048 + cg] + bhl[2048 + cg];
    const float zo = sZ[3][r][col] + xp_t[xb + 3072] + bxl[3072 + cg] + bhl[3072 + cg];
    const float i_ = 1.f / (1.f + __expf(-zi));
    const float f_ = 1.f / (1.f + __expf(-zf));
    const float g_ = tanhf(zg);
    const float o_ = 1.f / (1.f + __expf(-zo));
    const size_t sb = (size_t)r * H_DIM + cg;
    const float c  = f_ * c_state[sb] + i_ * g_;
    const float h  = o_ * tanhf(c);
    c_state[sb] = c;
    h_f32[sb]   = h;
    h_bf16[sb]  = f2bf(h);
  }
}

// ---------------- host orchestration ----------------

extern "C" void kernel_launch(void* const* d_in, const int* in_sizes, int n_in,
                              void* d_out, int out_size, void* d_ws, size_t ws_size,
                              hipStream_t stream) {
  const float* x  = (const float*)d_in[0];
  const float* h0 = (const float*)d_in[1];
  const float* c0 = (const float*)d_in[2];
  const float* Wx = (const float*)d_in[3];
  const float* Wh = (const float*)d_in[4];
  const float* bx = (const float*)d_in[5];
  const float* bh = (const float*)d_in[6];

  float* out_seq = (float*)d_out;                                  // (T,B,H)
  float* out_hs  = out_seq + (size_t)T_DIM * B_DIM * H_DIM;        // (L,B,H)
  float* out_cs  = out_hs + (size_t)L_DIM * B_DIM * H_DIM;         // (L,B,H)

  char* ws = (char*)d_ws;
  u16* x_bf   = (u16*)ws;   ws += (size_t)T_DIM * B_DIM * H_DIM * 2;  // 67 MB
  u16* wx_bf  = (u16*)ws;   ws += (size_t)G4 * H_DIM * 2;             // 8 MB
  u16* wh_bf  = (u16*)ws;   ws += (size_t)G4 * H_DIM * 2;             // 8 MB
  float* xp   = (float*)ws; ws += (size_t)TC * B_DIM * G4 * 4;        // 67 MB
  u16* h0_bf  = (u16*)ws;   ws += (size_t)B_DIM * H_DIM * 2;
  float* c_st = (float*)ws; ws += (size_t)B_DIM * H_DIM * 4;

  cvt_f32_bf16<<<2048, 256, 0, stream>>>(x, x_bf, (size_t)T_DIM * B_DIM * H_DIM);

  for (int l = 0; l < L_DIM; ++l) {
    cvt_f32_bf16<<<2048, 256, 0, stream>>>(Wx + (size_t)l * G4 * H_DIM, wx_bf,
                                           (size_t)G4 * H_DIM);
    cvt_f32_bf16<<<2048, 256, 0, stream>>>(Wh + (size_t)l * G4 * H_DIM, wh_bf,
                                           (size_t)G4 * H_DIM);
    cvt_f32_bf16<<<256, 256, 0, stream>>>(h0 + (size_t)l * B_DIM * H_DIM, h0_bf,
                                          (size_t)B_DIM * H_DIM);
    copy_f32<<<256, 256, 0, stream>>>(c0 + (size_t)l * B_DIM * H_DIM, c_st,
                                      (size_t)B_DIM * H_DIM);

    for (int t0 = 0; t0 < T_DIM; t0 += TC) {
      lstm_gemm_xw<<<dim3(TC * B_DIM / 128, G4 / 128), 256, 0, stream>>>(
          (const u32*)(x_bf + (size_t)t0 * B_DIM * H_DIM), (const u32*)wx_bf, xp);
      for (int t = t0; t < t0 + TC; ++t) {
        const u16* hp = (t == 0) ? h0_bf : (x_bf + (size_t)(t - 1) * B_DIM * H_DIM);
        lstm_step<<<64, 256, 0, stream>>>(
            xp + (size_t)(t - t0) * B_DIM * G4, hp, (const u32*)wh_bf,
            bx + (size_t)l * G4, bh + (size_t)l * G4, c_st,
            out_seq + (size_t)t * B_DIM * H_DIM,
            x_bf + (size_t)t * B_DIM * H_DIM);
      }
    }
    copy_f32<<<256, 256, 0, stream>>>(out_seq + (size_t)(T_DIM - 1) * B_DIM * H_DIM,
                                      out_hs + (size_t)l * B_DIM * H_DIM,
                                      (size_t)B_DIM * H_DIM);
    copy_f32<<<256, 256, 0, stream>>>(c_st, out_cs + (size_t)l * B_DIM * H_DIM,
                                      (size_t)B_DIM * H_DIM);
  }
}
